// CrossAttention_68942815036031
// MI455X (gfx1250) — matrix operand order repaired and checked by reading, not compiled, not run on hardware
//
#include <hip/hip_runtime.h>
#include <hip/hip_bf16.h>
#include <math.h>

// ---------------------------------------------------------------------------
// CDNA5 (gfx1250) wave32 WMMA cross-attention implementation.
// All GEMMs use v_wmma_f32_16x16x32_f16 (f16 inputs, f32 accumulate).
// Weights are pre-converted once to f16 N-major (B-fragment) layout.
// Encoder-MLP weight slabs are staged into LDS with the Tensor Data Mover
// (double-buffered), waited with s_wait_tensorcnt.
// ---------------------------------------------------------------------------

typedef __attribute__((ext_vector_type(16))) _Float16 v16h;
typedef __attribute__((ext_vector_type(8)))  _Float16 v8h;
typedef __attribute__((ext_vector_type(8)))  float    v8f;
typedef __attribute__((ext_vector_type(4))) unsigned int v4u;
typedef __attribute__((ext_vector_type(8)))  int v8i;
typedef __attribute__((ext_vector_type(4)))  int v4i;

#define NGROUP 64
#define CDIM   256
#define INDIM  512
#define EMBDIM 512
#define NHEAD  8
#define HDIM   32
#define SPAD   512   // padded per-group S extent for transposed V

#if __has_builtin(__builtin_amdgcn_tensor_load_to_lds)
#define HAVE_TDM 1
#else
#define HAVE_TDM 0
#endif

__device__ __forceinline__ v8f wmma16(v16h a, v16h b, v8f c) {
  // D = A(16x32 f16) * B(32x16 f16) + C(16x16 f32)
  return __builtin_amdgcn_wmma_f32_16x16x32_f16(false, a, false, b,
                                                (short)0, c, false, false);
}

// A-fragment (16x32, f16): lane m = lane&15; lanes16-31 shift K by 16.
// halves 0..15 -> K = koff+0..15
__device__ __forceinline__ v16h ld_afrag(const _Float16* base, int ld) {
  int lane = threadIdx.x & 31;
  int m = lane & 15;
  int koff = (lane & 16) ? 16 : 0;
  const _Float16* p = base + (size_t)m * ld + koff;
  union { v16h v; v8h h[2]; } u;
  u.h[0] = *(const v8h*)(p);
  u.h[1] = *(const v8h*)(p + 8);
  return u.v;
}

// B-fragment (32x16, f16) from K-contiguous storage: lane n = lane&15,
// halves 0..15 -> K = koff + 0..15 with koff = (lane<16)?0:16.
__device__ __forceinline__ v16h ld_bfrag(const _Float16* base, int ld) {
  int lane = threadIdx.x & 31;
  int n = lane & 15;
  int koff = (lane & 16) ? 16 : 0;
  return *(const v16h*)(base + (size_t)n * ld + koff);
}

#if HAVE_TDM
// 2D TDM load: tile [rows][tilek] of f16, global row stride kfull halfs,
// into contiguous LDS [rows][tilek]. Descriptor per CDNA5 ISA ch.8.
// 6-arg builtin form: (g0, g1, g2, g3, g4, cpol) with zero-filled hi groups.
__device__ __forceinline__ void tdm_load_2d(const _Float16* gsrc, unsigned lds_off,
                                            int tilek, int rows, int kfull) {
  unsigned long long ga = (unsigned long long)(uintptr_t)gsrc;
  v4u g0;
  g0[0] = 1u;                                                  // count=1, user D#
  g0[1] = lds_off;                                             // lds_addr
  g0[2] = (unsigned)ga;                                        // global_addr[31:0]
  g0[3] = (unsigned)((ga >> 32) & 0x01ffffffu) | 0x80000000u;  // [56:32] | type=2
  v8i g1;
  g1[0] = 1 << 16;                                             // data_size=2B
  g1[1] = (tilek & 0xffff) << 16;                              // tensor_dim0 lo16
  g1[2] = (int)(((unsigned)tilek >> 16) | ((unsigned)(rows & 0xffff) << 16)); // dim0 hi / dim1 lo
  g1[3] = (int)(((unsigned)rows >> 16) | ((unsigned)(tilek & 0xffff) << 16)); // dim1 hi / tile_dim0
  g1[4] = rows & 0xffff;                                       // tile_dim1 (tile_dim2=0)
  g1[5] = kfull;                                               // tensor_dim0_stride lo32
  g1[6] = 0;
  g1[7] = 0;
  v4i z4 = {0, 0, 0, 0};
  v8i z8 = {0, 0, 0, 0, 0, 0, 0, 0};
  __builtin_amdgcn_tensor_load_to_lds(g0, g1, z4, z4, z8, 0);
}
#endif

// ---------------------------------------------------------------------------
// 0) One-shot weight convert+transpose: src [K][N] f32 -> dst [N][K] f16
// ---------------------------------------------------------------------------
__global__ __launch_bounds__(256) void wconv(const float* __restrict__ src,
                                             _Float16* __restrict__ dst,
                                             int K, int N) {
  int idx = blockIdx.x * 256 + threadIdx.x;
  if (idx < K * N) {
    int k = idx % K, n = idx / K;
    dst[idx] = (_Float16)src[(size_t)k * N + n];
  }
}

// ---------------------------------------------------------------------------
// 1) Segment starts (lower_bound over sorted batch ids)
// ---------------------------------------------------------------------------
__global__ void seg_starts(const int* __restrict__ bx, int nx,
                           const int* __restrict__ be, int ne,
                           int* __restrict__ sx, int* __restrict__ se) {
  int g = threadIdx.x;
  if (g <= NGROUP) {
    int lo = 0, hi = nx;
    while (lo < hi) { int mid = (lo + hi) >> 1; if (bx[mid] < g) lo = mid + 1; else hi = mid; }
    sx[g] = lo;
    lo = 0; hi = ne;
    while (lo < hi) { int mid = (lo + hi) >> 1; if (be[mid] < g) lo = mid + 1; else hi = mid; }
    se[g] = lo;
  }
}

// ---------------------------------------------------------------------------
// 2) Encoder MLP: h = LN(GELU(E @ W1 + b1) @ W2 + b2), 32 rows per block.
//    Weight slabs staged into LDS via TDM, double-buffered.
// ---------------------------------------------------------------------------
__global__ __launch_bounds__(256) void enc_mlp(
    const float* __restrict__ E,
    const _Float16* __restrict__ W1t,   // [512 n][512 k] f16
    const float* __restrict__ b1,
    const _Float16* __restrict__ W2t,   // [256 n][512 k] f16
    const float* __restrict__ b2,
    const float* __restrict__ lnw, const float* __restrict__ lnb,
    _Float16* __restrict__ H, int Ne) {
  __shared__ __align__(64) unsigned char smem[32 * 512 * 2 + 2 * 512 * 32 * 2];
  __shared__ float mu_s[32], rs_s[32];
  _Float16* e16 = (_Float16*)smem;                               // 32 KB A tile / G
  _Float16* slab0 = (_Float16*)(smem + 32 * 512 * 2);            // 32 KB
  _Float16* slab1 = (_Float16*)(smem + 32 * 512 * 2 + 512 * 32 * 2);
  float* hbuf = (float*)(smem + 32 * 512 * 2);                   // overlay slabs

  int tid = threadIdx.x;
  int wave = tid >> 5, lane = tid & 31;
  int n0 = lane & 15, mb = (lane & 16) ? 8 : 0;
  int row0 = blockIdx.x * 32;

  // load E tile, convert f32->f16 (clamp rows)
  for (int i = 0; i < 64; ++i) {
    int idx = tid + i * 256;
    int r = idx >> 9, c = idx & 511;
    int src = row0 + r; if (src > Ne - 1) src = Ne - 1;
    e16[idx] = (_Float16)E[(size_t)src * INDIM + c];
  }

  v8f zero8 = {0, 0, 0, 0, 0, 0, 0, 0};

  // ---- phase 1: G = GELU(E @ W1 + b1), N=512, K=512, wave -> 64 N cols ----
  v8f acc1[2][4];
#pragma unroll
  for (int mt = 0; mt < 2; ++mt)
#pragma unroll
    for (int nt = 0; nt < 4; ++nt) acc1[mt][nt] = zero8;
  int nbase1 = wave * 64;
  _Float16* sl[2] = { slab0, slab1 };
  int cur = 0;
#if HAVE_TDM
  if (wave == 0)
    tdm_load_2d(W1t, (unsigned)(uintptr_t)slab0, 32, 512, EMBDIM);
#endif
  for (int k0 = 0; k0 < INDIM; k0 += 32) {
#if HAVE_TDM
    if (wave == 0) __builtin_amdgcn_s_wait_tensorcnt(0);
    __syncthreads();
    if (wave == 0 && k0 + 32 < INDIM)
      tdm_load_2d(W1t + (k0 + 32), (unsigned)(uintptr_t)sl[cur ^ 1], 32, 512, EMBDIM);
#else
    __syncthreads();
    for (int i = 0; i < 64; ++i) {
      int idx = tid + i * 256;           // 512*32 halfs
      int n = idx >> 5, k = idx & 31;
      sl[cur][idx] = W1t[(size_t)n * EMBDIM + k0 + k];
    }
    __syncthreads();
#endif
    v16h a0 = ld_afrag(e16 + k0, 512);
    v16h a1 = ld_afrag(e16 + 16 * 512 + k0, 512);
#pragma unroll
    for (int nt = 0; nt < 4; ++nt) {
      v16h b = ld_bfrag(sl[cur] + (nbase1 + nt * 16) * 32, 32);
      acc1[0][nt] = wmma16(a0, b, acc1[0][nt]);
      acc1[1][nt] = wmma16(a1, b, acc1[1][nt]);
    }
    cur ^= 1;
  }
  __syncthreads();
  // bias + exact GELU, write G (f16) over e16
#pragma unroll
  for (int mt = 0; mt < 2; ++mt)
#pragma unroll
    for (int nt = 0; nt < 4; ++nt) {
      int n = nbase1 + nt * 16 + n0;
      float bb = b1[n];
#pragma unroll
      for (int f = 0; f < 8; ++f) {
        float v = acc1[mt][nt][f] + bb;
        float ge = 0.5f * v * (1.0f + erff(v * 0.70710678118654752f));
        e16[(mt * 16 + mb + f) * 512 + n] = (_Float16)ge;
      }
    }
  __syncthreads();

  // ---- phase 2: Hp = G @ W2 + b2, N=256, K=512, wave -> 32 N cols ----
  v8f acc2[2][2];
#pragma unroll
  for (int mt = 0; mt < 2; ++mt)
#pragma unroll
    for (int nt = 0; nt < 2; ++nt) acc2[mt][nt] = zero8;
  int nbase2 = wave * 32;
  cur = 0;
#if HAVE_TDM
  if (wave == 0)
    tdm_load_2d(W2t, (unsigned)(uintptr_t)slab0, 32, 256, EMBDIM);
#endif
  for (int k0 = 0; k0 < EMBDIM; k0 += 32) {
#if HAVE_TDM
    if (wave == 0) __builtin_amdgcn_s_wait_tensorcnt(0);
    __syncthreads();
    if (wave == 0 && k0 + 32 < EMBDIM)
      tdm_load_2d(W2t + (k0 + 32), (unsigned)(uintptr_t)sl[cur ^ 1], 32, 256, EMBDIM);
#else
    __syncthreads();
    for (int i = 0; i < 32; ++i) {
      int idx = tid + i * 256;           // 256*32 halfs
      int n = idx >> 5, k = idx & 31;
      sl[cur][idx] = W2t[(size_t)n * EMBDIM + k0 + k];
    }
    __syncthreads();
#endif
    v16h a0 = ld_afrag(e16 + k0, 512);
    v16h a1 = ld_afrag(e16 + 16 * 512 + k0, 512);
#pragma unroll
    for (int nt = 0; nt < 2; ++nt) {
      v16h b = ld_bfrag(sl[cur] + (nbase2 + nt * 16) * 32, 32);
      acc2[0][nt] = wmma16(a0, b, acc2[0][nt]);
      acc2[1][nt] = wmma16(a1, b, acc2[1][nt]);
    }
    cur ^= 1;
  }
  __syncthreads();
#pragma unroll
  for (int mt = 0; mt < 2; ++mt)
#pragma unroll
    for (int nt = 0; nt < 2; ++nt) {
      int n = nbase2 + nt * 16 + n0;
      float bb = b2[n];
#pragma unroll
      for (int f = 0; f < 8; ++f)
        hbuf[(mt * 16 + mb + f) * 256 + n] = acc2[mt][nt][f] + bb;
    }
  __syncthreads();
  if (tid < 32) {                        // LayerNorm stats per row
    float s = 0.f, ss = 0.f;
    for (int c = 0; c < 256; ++c) { float v = hbuf[tid * 256 + c]; s += v; ss += v * v; }
    float mu = s * (1.0f / 256.0f);
    float var = ss * (1.0f / 256.0f) - mu * mu;
    mu_s[tid] = mu;
    rs_s[tid] = rsqrtf(var + 1e-5f);
  }
  __syncthreads();
  {
    int c = tid;                         // 256 threads == 256 columns
    float w = lnw[c], bb = lnb[c];
    for (int r = 0; r < 32; ++r) {
      int row = row0 + r;
      if (row < Ne)
        H[(size_t)row * CDIM + c] =
            (_Float16)((hbuf[r * 256 + c] - mu_s[r]) * rs_s[r] * w + bb);
    }
  }
}

// ---------------------------------------------------------------------------
// 3) Generic [M,256] x [256,256] + bias GEMM. 64 rows/block, 8 waves x 32 N.
//    B-fragments straight from pre-transposed f16 weights (L2-resident).
// ---------------------------------------------------------------------------
template <bool A_F32, bool OUT_F16>
__global__ __launch_bounds__(256) void gemm256(
    const void* __restrict__ Ain, const _Float16* __restrict__ Wt,  // [256 n][256 k]
    const float* __restrict__ bias, void* __restrict__ Out, int M) {
  __shared__ __align__(64) _Float16 a16[64 * 256];   // 32 KB
  int tid = threadIdx.x, wave = tid >> 5, lane = tid & 31;
  int n0 = lane & 15, mb = (lane & 16) ? 8 : 0;
  int row0 = blockIdx.x * 64;
  const float*    Af = (const float*)Ain;
  const _Float16* Ah = (const _Float16*)Ain;
  for (int i = 0; i < 64; ++i) {
    int idx = tid + i * 256;
    int r = idx >> 8, c = idx & 255;
    int src = row0 + r; if (src > M - 1) src = M - 1;
    a16[idx] = A_F32 ? (_Float16)Af[(size_t)src * CDIM + c]
                     : Ah[(size_t)src * CDIM + c];
  }
  __syncthreads();
  v8f zero8 = {0, 0, 0, 0, 0, 0, 0, 0};
  v8f acc[4][2];
#pragma unroll
  for (int mt = 0; mt < 4; ++mt) { acc[mt][0] = zero8; acc[mt][1] = zero8; }
  int nbase = wave * 32;
  for (int k0 = 0; k0 < 256; k0 += 32) {
    v16h b0 = ld_bfrag(Wt + (size_t)nbase * 256 + k0, 256);
    v16h b1 = ld_bfrag(Wt + (size_t)(nbase + 16) * 256 + k0, 256);
#pragma unroll
    for (int mt = 0; mt < 4; ++mt) {
      v16h a = ld_afrag(a16 + mt * 16 * 256 + k0, 256);
      acc[mt][0] = wmma16(a, b0, acc[mt][0]);
      acc[mt][1] = wmma16(a, b1, acc[mt][1]);
    }
  }
#pragma unroll
  for (int mt = 0; mt < 4; ++mt)
#pragma unroll
    for (int nt = 0; nt < 2; ++nt) {
      int n = nbase + nt * 16 + n0;
      float bb = bias[n];
#pragma unroll
      for (int f = 0; f < 8; ++f) {
        int row = row0 + mt * 16 + mb + f;
        if (row < M) {
          float v = acc[mt][nt][f] + bb;
          if (OUT_F16) ((_Float16*)Out)[(size_t)row * CDIM + n] = (_Float16)v;
          else         ((float*)Out)[(size_t)row * CDIM + n] = v;
        }
      }
    }
}

// ---------------------------------------------------------------------------
// 4) Transpose V into per-group [c][SPAD] layout (zero-padded tail).
// ---------------------------------------------------------------------------
__global__ __launch_bounds__(256) void transpose_v(
    const _Float16* __restrict__ V, const int* __restrict__ se,
    _Float16* __restrict__ VT) {
  int idx = blockIdx.x * 256 + threadIdx.x;  // NGROUP * 256 * SPAD threads
  int s = idx & (SPAD - 1);
  int c = (idx >> 9) & 255;
  int g = idx >> 17;
  int es = se[g];
  int Sg = se[g + 1] - es;
  _Float16 val = (_Float16)0.0f;
  if (s < Sg) val = V[(size_t)(es + s) * CDIM + c];
  VT[idx] = val;
}

// ---------------------------------------------------------------------------
// 5) Flash attention: one wave per 16-query tile per (group, head).
// ---------------------------------------------------------------------------
__global__ __launch_bounds__(256) void attn_kernel(
    const _Float16* __restrict__ Q, const _Float16* __restrict__ K,
    const _Float16* __restrict__ VT, const int* __restrict__ Mask,
    _Float16* __restrict__ Y, const int* __restrict__ sx,
    const int* __restrict__ se) {
  __shared__ __align__(32) _Float16 scbuf[8][16 * 32];   // per-wave P patch
  int gb = blockIdx.x;
  int g = gb >> 3;
  int head = gb & 7;
  int wave = threadIdx.x >> 5, lane = threadIdx.x & 31;
  int xs = sx[g], Tg = sx[g + 1] - xs;
  int es = se[g], Sg = se[g + 1] - es;
  int t0 = blockIdx.y * 128 + wave * 16;
  if (t0 >= Tg) return;                 // wave-uniform; no block barriers here
  _Float16* sc = scbuf[wave];

  int n0 = lane & 15, mb = (lane & 16) ? 8 : 0;

  v16h qa = ld_afrag(Q + (size_t)(xs + t0) * CDIM + head * HDIM, CDIM);

  v8f o0 = {0, 0, 0, 0, 0, 0, 0, 0}, o1 = {0, 0, 0, 0, 0, 0, 0, 0};
  float mrow[8], lrow[8];
#pragma unroll
  for (int f = 0; f < 8; ++f) { mrow[f] = -3.0e38f; lrow[f] = 0.0f; }

  const float scale = 0.17677669529663687f;   // 1/sqrt(HD)
  const _Float16* vtg = VT + ((size_t)g * CDIM + head * HDIM) * SPAD;

  for (int s0 = 0; s0 < Sg; s0 += 32) {
    v16h kb0 = ld_bfrag(K + (size_t)(es + s0) * CDIM + head * HDIM, CDIM);
    v16h kb1 = ld_bfrag(K + (size_t)(es + s0 + 16) * CDIM + head * HDIM, CDIM);
    v8f z = {0, 0, 0, 0, 0, 0, 0, 0};
    v8f st0 = wmma16(qa, kb0, z);
    v8f st1 = wmma16(qa, kb1, z);

    int sA = s0 + n0, sB = s0 + 16 + n0;
    int mk0 = (sA < Sg) ? Mask[es + sA] : 0;
    int mk1 = (sB < Sg) ? Mask[es + sB] : 0;

#pragma unroll
    for (int f = 0; f < 8; ++f) {
      float e0 = mk0 ? st0[f] * scale : -3.0e38f;
      float e1 = mk1 ? st1[f] * scale : -3.0e38f;
      float mx = fmaxf(e0, e1);
      mx = fmaxf(mx, __shfl_xor(mx, 1));
      mx = fmaxf(mx, __shfl_xor(mx, 2));
      mx = fmaxf(mx, __shfl_xor(mx, 4));
      mx = fmaxf(mx, __shfl_xor(mx, 8));
      float mn = fmaxf(mrow[f], mx);
      float corr = __expf(mrow[f] - mn);
      mrow[f] = mn;
      float p0 = __expf(e0 - mn);
      float p1 = __expf(e1 - mn);
      float ps = p0 + p1;
      ps += __shfl_xor(ps, 1);
      ps += __shfl_xor(ps, 2);
      ps += __shfl_xor(ps, 4);
      ps += __shfl_xor(ps, 8);
      lrow[f] = lrow[f] * corr + ps;
      o0[f] *= corr;
      o1[f] *= corr;
      sc[(mb + f) * 32 + n0]      = (_Float16)p0;   // D-layout -> LDS patch
      sc[(mb + f) * 32 + 16 + n0] = (_Float16)p1;
    }
    // wave-synchronous LDS round-trip: order DS reads after DS writes
    asm volatile("s_wait_dscnt 0" ::: "memory");
    v16h pa  = ld_afrag(sc, 32);
    v16h vb0 = ld_bfrag(vtg + s0, SPAD);
    v16h vb1 = ld_bfrag(vtg + 16 * SPAD + s0, SPAD);
    o0 = wmma16(pa, vb0, o0);
    o1 = wmma16(pa, vb1, o1);
  }

#pragma unroll
  for (int f = 0; f < 8; ++f) {
    int t = t0 + mb + f;
    if (t < Tg) {
      float inv = 1.0f / lrow[f];
      size_t base = (size_t)(xs + t) * CDIM + head * HDIM + n0;
      Y[base]      = (_Float16)(o0[f] * inv);
      Y[base + 16] = (_Float16)(o1[f] * inv);
    }
  }
}

// ---------------------------------------------------------------------------
// Host-side launch
// ---------------------------------------------------------------------------
extern "C" void kernel_launch(void* const* d_in, const int* in_sizes, int n_in,
                              void* d_out, int out_size, void* d_ws, size_t ws_size,
                              hipStream_t stream) {
  (void)n_in; (void)out_size; (void)ws_size;
  const float* x    = (const float*)d_in[0];
  const int*   bx   = (const int*)d_in[1];
  const int*   be   = (const int*)d_in[2];
  const float* ee   = (const float*)d_in[3];
  const int*   mask = (const int*)d_in[4];
  const float* Wq = (const float*)d_in[5];  const float* bq = (const float*)d_in[6];
  const float* Wk = (const float*)d_in[7];  const float* bk = (const float*)d_in[8];
  const float* Wv = (const float*)d_in[9];  const float* bv = (const float*)d_in[10];
  const float* Wp = (const float*)d_in[11]; const float* bp = (const float*)d_in[12];
  const float* W1 = (const float*)d_in[13]; const float* b1 = (const float*)d_in[14];
  const float* W2 = (const float*)d_in[15]; const float* b2 = (const float*)d_in[16];
  const float* lnw = (const float*)d_in[17]; const float* lnb = (const float*)d_in[18];

  int Nx = in_sizes[1];
  int Ne = in_sizes[2];
  size_t NxP = ((size_t)Nx + 63) & ~(size_t)63;
  size_t NeP = ((size_t)Ne + 63) & ~(size_t)63;

  unsigned char* ws = (unsigned char*)d_ws;
  size_t off = 0;
  auto take = [&](size_t bytes) {
    unsigned char* p = ws + off;
    off = (off + bytes + 255) & ~(size_t)255;
    return p;
  };
  int* sx = (int*)take((NGROUP + 1) * sizeof(int));
  int* se = (int*)take((NGROUP + 1) * sizeof(int));
  _Float16* w1t = (_Float16*)take((size_t)512 * 512 * 2);
  _Float16* w2t = (_Float16*)take((size_t)256 * 512 * 2);
  _Float16* wqt = (_Float16*)take((size_t)256 * 256 * 2);
  _Float16* wkt = (_Float16*)take((size_t)256 * 256 * 2);
  _Float16* wvt = (_Float16*)take((size_t)256 * 256 * 2);
  _Float16* wpt = (_Float16*)take((size_t)256 * 256 * 2);
  _Float16* hf = (_Float16*)take(NeP * CDIM * 2);
  _Float16* qf = (_Float16*)take(NxP * CDIM * 2);
  _Float16* kf = (_Float16*)take(NeP * CDIM * 2);
  _Float16* vf = (_Float16*)take(NeP * CDIM * 2);
  _Float16* vt = (_Float16*)take((size_t)NGROUP * CDIM * SPAD * 2);
  _Float16* yf = (_Float16*)take(NxP * CDIM * 2);

  // one-shot weight convert + transpose to f16 B-fragment layout
  wconv<<<(512 * 512 + 255) / 256, 256, 0, stream>>>(W1, w1t, 512, 512);
  wconv<<<(512 * 256 + 255) / 256, 256, 0, stream>>>(W2, w2t, 512, 256);
  wconv<<<(256 * 256 + 255) / 256, 256, 0, stream>>>(Wq, wqt, 256, 256);
  wconv<<<(256 * 256 + 255) / 256, 256, 0, stream>>>(Wk, wkt, 256, 256);
  wconv<<<(256 * 256 + 255) / 256, 256, 0, stream>>>(Wv, wvt, 256, 256);
  wconv<<<(256 * 256 + 255) / 256, 256, 0, stream>>>(Wp, wpt, 256, 256);

  seg_starts<<<1, 128, 0, stream>>>(bx, Nx, be, Ne, sx, se);
  enc_mlp<<<(Ne + 31) / 32, 256, 0, stream>>>(ee, w1t, b1, w2t, b2, lnw, lnb, hf, Ne);
  gemm256<true,  true ><<<(Nx + 63) / 64, 256, 0, stream>>>(x,  wqt, bq, qf, Nx);
  gemm256<false, true ><<<(Ne + 63) / 64, 256, 0, stream>>>(hf, wkt, bk, kf, Ne);
  gemm256<false, true ><<<(Ne + 63) / 64, 256, 0, stream>>>(hf, wvt, bv, vf, Ne);
  transpose_v<<<NGROUP * SPAD, 256, 0, stream>>>(vf, se, vt);
  dim3 ag(NGROUP * NHEAD, 4);
  attn_kernel<<<ag, 256, 0, stream>>>(qf, kf, vt, mask, yf, sx, se);
  gemm256<false, false><<<(Nx + 63) / 64, 256, 0, stream>>>(yf, wpt, bp, d_out, Nx);
}
